// LSTM_W2V_Net_Cnn_Preload_34239479284276
// MI455X (gfx1250) — compile-verified, matrix operand order checked by reading, not run on hardware
//
#include <hip/hip_runtime.h>
#include <hip/hip_bf16.h>

// ---------------------------------------------------------------------------
// LSTM decoder for MI455X (gfx1250, wave32).
// All GEMMs use v_wmma_f32_16x16x32_bf16 (bf16 inputs, f32 accumulate) with
// A-fragment register reuse across multiple N-tiles per wave.
// Packed layouts (K padded to multiples of 32):
//   XH   : 32 slots of (64 x 832) bf16 = [ Xp(256) | e(300)+pad(20) | h(256) ]
//   Wcat : (1024 x 832) bf16          = [ w_ih[:, :556] | 0-pad | w_hh ]
// Step t: gates GEMM reads XH[t]; the per-lane LSTM pointwise (no LDS, no
// barriers -- each lane owns i,f,g,o of its elements in registers) writes h
// into XH[t+1]'s h-region; fc2 reads XH[t+1]; log-softmax runs in place.
// ---------------------------------------------------------------------------

typedef __bf16 bf16_t;
typedef __attribute__((ext_vector_type(16))) __bf16 v16bf;
typedef __attribute__((ext_vector_type(8)))  __bf16 v8bf;
typedef __attribute__((ext_vector_type(8)))  float  v8f;

#define BATCH 64
#define HID   256
#define EMB   300
#define EMBP  320      // padded embedding width
#define KCAT  832      // 256 + 320 + 256, multiple of 32
#define HOFF  576      // offset of h-region inside an XH row
#define VOC   10000
#define FEAT  4096
#define TSTEP 32
#define NSTEP 31
#define SLOT  (BATCH * KCAT)          // elements per XH slot
#define OUTLD ((size_t)TSTEP * VOC)   // row stride (per batch) in d_out

// --- bf16 fragment load following CDNA5 16-bit A/B layout ------------------
// lane half h: element i in {0..7} -> k = i + 8h ; i in {8..15} -> k = 16+i-8+8h
__device__ __forceinline__ v16bf load_frag(const bf16_t* __restrict__ p) {
  v8bf a = *(const v8bf*)(p);        // k = 0..7   (+8*half already in p)
  v8bf b = *(const v8bf*)(p + 16);   // k = 16..23 (+8*half)
  return __builtin_shufflevector(a, b, 0,1,2,3,4,5,6,7,8,9,10,11,12,13,14,15);
}

// per-lane base pointer of an A/B fragment whose rows start at r0
__device__ __forceinline__ const bf16_t* frag_base(const bf16_t* __restrict__ M,
                                                   int ld, int r0) {
  const int lane = threadIdx.x & 31;
  return M + (size_t)(r0 + (lane & 15)) * ld + 8 * (lane >> 4);
}

__device__ __forceinline__ v8f wmma_bf16(v16bf a, v16bf b, v8f c) {
  return __builtin_amdgcn_wmma_f32_16x16x32_bf16(false, a, false, b,
                                                 (short)0, c, false, false);
}

__device__ __forceinline__ float sigmoidf_(float x) {
  return 1.f / (1.f + __expf(-x));
}

// --------------------------- setup kernels ---------------------------------

__global__ void k_f32_to_bf16(const float* __restrict__ s, bf16_t* __restrict__ d, int n) {
  int i = blockIdx.x * blockDim.x + threadIdx.x;
  if (i < n) d[i] = (bf16_t)s[i];
}

__global__ void k_build_wcat(const float* __restrict__ w_ih, const float* __restrict__ w_hh,
                             bf16_t* __restrict__ Wcat) {
  int i = blockIdx.x * blockDim.x + threadIdx.x;
  if (i >= 1024 * KCAT) return;
  int n = i / KCAT, k = i % KCAT;
  float v;
  if (k < 556)       v = w_ih[n * 556 + k];          // x|e part (e cols 256..555)
  else if (k < HOFF) v = 0.f;                        // pad 556..575
  else               v = w_hh[n * HID + (k - HOFF)]; // h part
  Wcat[i] = (bf16_t)v;
}

__global__ void k_bcat(const float* __restrict__ bi, const float* __restrict__ bh,
                       float* __restrict__ bc) {
  int i = blockIdx.x * blockDim.x + threadIdx.x;
  if (i < 1024) bc[i] = bi[i] + bh[i];
}

// fill e-region of slots 0..30 : slot 0 = emb[START_ID], slot t = emb[label[t]]
__global__ void k_embed(const float* __restrict__ emb, const int* __restrict__ label,
                        bf16_t* __restrict__ XH) {
  int i = blockIdx.x * blockDim.x + threadIdx.x;
  if (i >= NSTEP * BATCH * EMBP) return;
  int t = i / (BATCH * EMBP);
  int r = i % (BATCH * EMBP);
  int b = r / EMBP, j = r % EMBP;
  float v = 0.f;
  if (j < EMB) {
    int id = (t == 0) ? 1 : label[t * BATCH + b];    // START_ID = 1
    v = emb[(size_t)id * EMB + j];
  }
  XH[(size_t)t * SLOT + (size_t)b * KCAT + HID + j] = (bf16_t)v;
}

__global__ void k_init_state(float* __restrict__ C, bf16_t* __restrict__ XH0) {
  int i = blockIdx.x * blockDim.x + threadIdx.x;
  if (i >= BATCH * HID) return;
  C[i] = 0.f;
  int b = i >> 8, j = i & 255;
  XH0[(size_t)b * KCAT + HOFF + j] = (bf16_t)0.f;    // h0 = 0
}

__global__ void k_onehot(float* __restrict__ out) {   // out[:, 0, :] = onehot(START_ID)
  int i = blockIdx.x * blockDim.x + threadIdx.x;
  if (i >= BATCH * VOC) return;
  int b = i / VOC, v = i % VOC;
  out[(size_t)b * OUTLD + v] = (v == 1) ? 1.f : 0.f;
}

// fc1: Xp = X @ fc1_w^T + b, written (bf16) into Xp-region of XH slots 0..30.
// One wave owns 4 N-tiles, A fragment reused across them. grid(4,1), block 128.
__global__ void k_fc1(const bf16_t* __restrict__ Xb, const bf16_t* __restrict__ W1,
                      const float* __restrict__ b1, bf16_t* __restrict__ XH) {
  int wave = threadIdx.x >> 5, lane = threadIdx.x & 31;
  int m0 = blockIdx.x * 16;
  int n0 = wave * 64;
  const bf16_t* ar = frag_base(Xb, FEAT, m0);
  const bf16_t* w0 = frag_base(W1, FEAT, n0);
  const bf16_t* w1 = frag_base(W1, FEAT, n0 + 16);
  const bf16_t* w2 = frag_base(W1, FEAT, n0 + 32);
  const bf16_t* w3 = frag_base(W1, FEAT, n0 + 48);
  v8f c0 = {}, c1 = {}, c2 = {}, c3 = {};
  for (int k = 0; k < FEAT; k += 32) {
    v16bf a = load_frag(ar + k);
    c0 = wmma_bf16(a, load_frag(w0 + k), c0);
    c1 = wmma_bf16(a, load_frag(w1 + k), c1);
    c2 = wmma_bf16(a, load_frag(w2 + k), c2);
    c3 = wmma_bf16(a, load_frag(w3 + k), c3);
  }
  int half = lane >> 4, col = lane & 15;
#pragma unroll
  for (int r = 0; r < 8; r++) {
    int mg = m0 + r + 8 * half;
    float v0 = c0[r] + b1[n0 + col];
    float v1 = c1[r] + b1[n0 + 16 + col];
    float v2 = c2[r] + b1[n0 + 32 + col];
    float v3 = c3[r] + b1[n0 + 48 + col];
    for (int t = 0; t < NSTEP; t++) {
      bf16_t* row = XH + (size_t)t * SLOT + (size_t)mg * KCAT;
      row[n0 + col]      = (bf16_t)v0;
      row[n0 + 16 + col] = (bf16_t)v1;
      row[n0 + 32 + col] = (bf16_t)v2;
      row[n0 + 48 + col] = (bf16_t)v3;
    }
  }
}

// --------------------------- per-step kernels ------------------------------

// One wave computes i,f,g,o tiles for its own 16x16 (b,j) tile: A fragment
// loaded once per K-slice, fed to 4 WMMAs; pointwise fully in registers
// (each lane owns all four gate values of its elements). grid(4,4), block 128.
__global__ void k_lstm_step(const bf16_t* __restrict__ XHt, bf16_t* __restrict__ XHn,
                            const bf16_t* __restrict__ Wcat, const float* __restrict__ bcat,
                            float* __restrict__ Cst) {
  int wave = threadIdx.x >> 5, lane = threadIdx.x & 31;
  int m0 = blockIdx.x * 16;                 // batch tile
  int j0 = blockIdx.y * 64 + wave * 16;     // hidden tile (0..240)
  const bf16_t* ar = frag_base(XHt, KCAT, m0);
  const bf16_t* wi = frag_base(Wcat, KCAT, 0 * HID + j0);
  const bf16_t* wf = frag_base(Wcat, KCAT, 1 * HID + j0);
  const bf16_t* wg = frag_base(Wcat, KCAT, 2 * HID + j0);
  const bf16_t* wo = frag_base(Wcat, KCAT, 3 * HID + j0);
  v8f ci = {}, cf = {}, cg = {}, co = {};
  for (int k = 0; k < KCAT; k += 32) {
    v16bf a = load_frag(ar + k);
    ci = wmma_bf16(a, load_frag(wi + k), ci);
    cf = wmma_bf16(a, load_frag(wf + k), cf);
    cg = wmma_bf16(a, load_frag(wg + k), cg);
    co = wmma_bf16(a, load_frag(wo + k), co);
  }
  int half = lane >> 4, col = lane & 15;
  int jg = j0 + col;
  float bi = bcat[0 * HID + jg], bff = bcat[1 * HID + jg];
  float bg = bcat[2 * HID + jg], bo = bcat[3 * HID + jg];
#pragma unroll
  for (int r = 0; r < 8; r++) {
    int mg = m0 + r + 8 * half;
    float iv = sigmoidf_(ci[r] + bi);
    float fv = sigmoidf_(cf[r] + bff);
    float gv = tanhf(cg[r] + bg);
    float ov = sigmoidf_(co[r] + bo);
    float c = fv * Cst[mg * HID + jg] + iv * gv;
    Cst[mg * HID + jg] = c;
    float h = ov * tanhf(c);
    XHn[(size_t)mg * KCAT + HOFF + jg] = (bf16_t)h;   // h for step t+1 and fc2
  }
}

// logits = h @ fc2_w^T + b, raw into d_out[:, t, :].
// One wave owns 5 N-tiles (80 cols); 125 wave-slots cover V=10000 exactly.
// grid(4,32), block 128 (slots >= 125 retire uniformly).
__global__ void k_fc2(const bf16_t* __restrict__ XHn, const bf16_t* __restrict__ W2,
                      const float* __restrict__ bias2, float* __restrict__ out) {
  int wave = threadIdx.x >> 5, lane = threadIdx.x & 31;
  int slot = blockIdx.y * 4 + wave;
  if (slot >= 125) return;                  // wave-uniform, EXEC stays all-ones
  int m0 = blockIdx.x * 16;
  int n0 = slot * 80;
  const bf16_t* ar = frag_base(XHn + HOFF, KCAT, m0);
  const bf16_t* w0 = frag_base(W2, HID, n0);
  const bf16_t* w1 = frag_base(W2, HID, n0 + 16);
  const bf16_t* w2 = frag_base(W2, HID, n0 + 32);
  const bf16_t* w3 = frag_base(W2, HID, n0 + 48);
  const bf16_t* w4 = frag_base(W2, HID, n0 + 64);
  v8f c0 = {}, c1 = {}, c2 = {}, c3 = {}, c4 = {};
  for (int k = 0; k < HID; k += 32) {
    v16bf a = load_frag(ar + k);
    c0 = wmma_bf16(a, load_frag(w0 + k), c0);
    c1 = wmma_bf16(a, load_frag(w1 + k), c1);
    c2 = wmma_bf16(a, load_frag(w2 + k), c2);
    c3 = wmma_bf16(a, load_frag(w3 + k), c3);
    c4 = wmma_bf16(a, load_frag(w4 + k), c4);
  }
  int half = lane >> 4, col = lane & 15;
  float b0 = bias2[n0 + col],      b1 = bias2[n0 + 16 + col];
  float b2 = bias2[n0 + 32 + col], b3 = bias2[n0 + 48 + col];
  float b4 = bias2[n0 + 64 + col];
#pragma unroll
  for (int r = 0; r < 8; r++) {
    float* row = out + (size_t)(m0 + r + 8 * half) * OUTLD + n0 + col;
    row[0]  = c0[r] + b0;
    row[16] = c1[r] + b1;
    row[32] = c2[r] + b2;
    row[48] = c3[r] + b3;
    row[64] = c4[r] + b4;
  }
}

// fused log_softmax over one output row (10000), in place in d_out
__global__ void k_logsoftmax(float* __restrict__ out) {
  __shared__ float red[256];
  float* row = out + (size_t)blockIdx.x * OUTLD;
  float mx = -INFINITY;
  for (int v = threadIdx.x; v < VOC; v += 256) mx = fmaxf(mx, row[v]);
  red[threadIdx.x] = mx; __syncthreads();
  for (int s = 128; s > 0; s >>= 1) {
    if (threadIdx.x < s) red[threadIdx.x] = fmaxf(red[threadIdx.x], red[threadIdx.x + s]);
    __syncthreads();
  }
  mx = red[0]; __syncthreads();
  float sum = 0.f;
  for (int v = threadIdx.x; v < VOC; v += 256) sum += __expf(row[v] - mx);
  red[threadIdx.x] = sum; __syncthreads();
  for (int s = 128; s > 0; s >>= 1) {
    if (threadIdx.x < s) red[threadIdx.x] += red[threadIdx.x + s];
    __syncthreads();
  }
  float lse = mx + __logf(red[0]);
  __syncthreads();
  for (int v = threadIdx.x; v < VOC; v += 256) row[v] -= lse;
}

// ---------------------------------------------------------------------------

extern "C" void kernel_launch(void* const* d_in, const int* in_sizes, int n_in,
                              void* d_out, int out_size, void* d_ws, size_t ws_size,
                              hipStream_t stream) {
  const float* X      = (const float*)d_in[0];
  const float* emb    = (const float*)d_in[1];
  const float* fc1_w  = (const float*)d_in[2];
  const float* fc1_b  = (const float*)d_in[3];
  const float* w_ih   = (const float*)d_in[4];
  const float* w_hh   = (const float*)d_in[5];
  const float* b_ih   = (const float*)d_in[6];
  const float* b_hh   = (const float*)d_in[7];
  const float* fc2_w  = (const float*)d_in[8];
  const float* fc2_b  = (const float*)d_in[9];
  const int*   label  = (const int*)d_in[10];
  float* out = (float*)d_out;

  // workspace layout (256B aligned)
  char* ws = (char*)d_ws;
  size_t off = 0;
  auto alloc = [&](size_t bytes) { void* p = ws + off; off = (off + bytes + 255) & ~(size_t)255; return p; };
  bf16_t* XH   = (bf16_t*)alloc((size_t)TSTEP * SLOT * sizeof(bf16_t));   // 3.4 MB
  bf16_t* Wcat = (bf16_t*)alloc((size_t)1024 * KCAT * sizeof(bf16_t));    // 1.7 MB
  bf16_t* Xb   = (bf16_t*)alloc((size_t)BATCH * FEAT * sizeof(bf16_t));   // 0.5 MB
  bf16_t* W1b  = (bf16_t*)alloc((size_t)HID * FEAT * sizeof(bf16_t));     // 2.1 MB
  bf16_t* W2b  = (bf16_t*)alloc((size_t)VOC * HID * sizeof(bf16_t));      // 5.1 MB
  float*  bcat = (float*)alloc(1024 * sizeof(float));
  float*  Cst  = (float*)alloc((size_t)BATCH * HID * sizeof(float));

  // one-time packing / conversion
  k_f32_to_bf16<<<(BATCH * FEAT + 255) / 256, 256, 0, stream>>>(X, Xb, BATCH * FEAT);
  k_f32_to_bf16<<<(HID * FEAT + 255) / 256, 256, 0, stream>>>(fc1_w, W1b, HID * FEAT);
  k_f32_to_bf16<<<(VOC * HID + 255) / 256, 256, 0, stream>>>(fc2_w, W2b, VOC * HID);
  k_build_wcat<<<(1024 * KCAT + 255) / 256, 256, 0, stream>>>(w_ih, w_hh, Wcat);
  k_bcat<<<4, 256, 0, stream>>>(b_ih, b_hh, bcat);
  k_embed<<<(NSTEP * BATCH * EMBP + 255) / 256, 256, 0, stream>>>(emb, label, XH);
  k_init_state<<<64, 256, 0, stream>>>(Cst, XH);
  k_fc1<<<dim3(4, 1), 128, 0, stream>>>(Xb, W1b, fc1_b, XH);
  k_onehot<<<(BATCH * VOC + 255) / 256, 256, 0, stream>>>(out);

  // 31 sequential decoder steps
  for (int t = 0; t < NSTEP; t++) {
    k_lstm_step<<<dim3(4, 4), 128, 0, stream>>>(XH + (size_t)t * SLOT,
                                                XH + (size_t)(t + 1) * SLOT,
                                                Wcat, bcat, Cst);
    k_fc2<<<dim3(4, 32), 128, 0, stream>>>(XH + (size_t)(t + 1) * SLOT,
                                           W2b, fc2_b,
                                           out + (size_t)(t + 1) * VOC);
    k_logsoftmax<<<64, 256, 0, stream>>>(out + (size_t)(t + 1) * VOC);
  }
}
